// VisionTransformer_21363167330619
// MI455X (gfx1250) — matrix operand order repaired and checked by reading, not compiled, not run on hardware
//
#include <hip/hip_runtime.h>
#include <hip/hip_bf16.h>
#include <math.h>

// ---------------------------------------------------------------------------
// Types / helpers
// ---------------------------------------------------------------------------
typedef __attribute__((ext_vector_type(16))) __bf16 v16bf;
typedef __attribute__((ext_vector_type(8)))  float  v8f;

__device__ __forceinline__ unsigned short f2bf(float f) {
    unsigned u = __float_as_uint(f);
    u += 0x7FFFu + ((u >> 16) & 1u);   // round-to-nearest-even
    return (unsigned short)(u >> 16);
}
__device__ __forceinline__ float bf2f(unsigned short h) {
    return __uint_as_float(((unsigned)h) << 16);
}

union BfFrag { v16bf v; unsigned u[8]; };

static constexpr int EPI_F32_BIAS      = 0;  // C = acc + bias          (f32 out)
static constexpr int EPI_F32_BIAS_RES  = 1;  // C = acc + bias + res    (f32 out)
static constexpr int EPI_BF16_BIAS     = 2;  // C = bf16(acc + bias)
static constexpr int EPI_BF16_BIAS_GEL = 3;  // C = bf16(gelu(acc+bias))
static constexpr int EPI_BF16_NOBIAS   = 4;  // C = bf16(acc)

// ---------------------------------------------------------------------------
// bf16 WMMA GEMM. 128 threads = 4 waves; each wave owns a 32(M)x64(N) tile:
//   per K-step: 2 A fragments (reused 4x) + 4 B fragments (reused 2x)
//   -> 8 x v_wmma_f32_16x16x32_bf16, 12 b128-loads  (1.5 loads / WMMA)
// Block covers 128(M) x 64(N).
// BT=true : B is [N][K] row-major (weights, K contiguous -> dword pair loads)
// BT=false: B is [K][N] with row stride ldb (attention V)
// Batched via blockIdx.z decomposed as z = zb*nh + zh with split strides.
// ---------------------------------------------------------------------------
template<int EPI, bool BT>
__global__ __launch_bounds__(128) void gemm_wmma(
    const unsigned short* __restrict__ A, int lda, long long sAb, long long sAh,
    const unsigned short* __restrict__ Bm, int ldb, long long sBb, long long sBh,
    const float* __restrict__ bias,
    const float* __restrict__ res,
    void* __restrict__ Cout, int ldc, long long sCb, long long sCh,
    int K, int nh)
{
    const int z  = blockIdx.z;
    const int zb = z / nh, zh = z % nh;
    A  += (size_t)zb * sAb + (size_t)zh * sAh;
    Bm += (size_t)zb * sBb + (size_t)zh * sBh;
    const size_t coff = (size_t)zb * sCb + (size_t)zh * sCh;

    const int lane = threadIdx.x & 31;
    const int wave = threadIdx.x >> 5;
    const int half = lane >> 4;      // 0: lanes 0-15, 1: lanes 16-31
    const int loc  = lane & 15;

    const int m0 = blockIdx.y * 128 + wave * 32;
    const int n0 = blockIdx.x * 64;

    v8f acc[2][4] = {};

    const unsigned short* Arow0 = A + (size_t)(m0 +      loc) * lda;
    const unsigned short* Arow1 = A + (size_t)(m0 + 16 + loc) * lda;

    for (int k0 = 0; k0 < K; k0 += 32) {
        // ---- two A fragments: 16x32 bf16 each, contiguous k = 16*half + s order --------
        BfFrag a0, a1;
#pragma unroll
        for (int r = 0; r < 8; ++r) {
            const int kb = 2*r + 16*half;
            a0.u[r] = *(const unsigned*)(Arow0 + k0 + kb);
            a1.u[r] = *(const unsigned*)(Arow1 + k0 + kb);
        }
        __builtin_prefetch(Arow0 + k0 + 64, 0, 3);   // near-scope prefetch
        __builtin_prefetch(Arow1 + k0 + 64, 0, 3);

        // ---- 4 B fragments (32x16 each), each feeds 2 WMMAs -------------
#pragma unroll
        for (int nt = 0; nt < 4; ++nt) {
            BfFrag bfr;
            if (BT) {
                const unsigned short* Brow =
                    Bm + (size_t)(n0 + nt*16 + loc) * ldb + k0;
#pragma unroll
                for (int r = 0; r < 8; ++r)
                    bfr.u[r] = *(const unsigned*)(Brow + 2*r + 16*half);
            } else {
                const int col = n0 + nt*16 + loc;
#pragma unroll
                for (int r = 0; r < 8; ++r) {
                    const int k = k0 + 2*r + 16*half;
                    unsigned lo = Bm[(size_t)k       * ldb + col];
                    unsigned hi = Bm[(size_t)(k + 1) * ldb + col];
                    bfr.u[r] = lo | (hi << 16);
                }
            }
            acc[0][nt] = __builtin_amdgcn_wmma_f32_16x16x32_bf16(
                false, a0.v, false, bfr.v, (short)0, acc[0][nt], false, false);
            acc[1][nt] = __builtin_amdgcn_wmma_f32_16x16x32_bf16(
                false, a1.v, false, bfr.v, (short)0, acc[1][nt], false, false);
        }
    }

    // ---- epilogue + store: C VGPR j / lane -> (j + 8*half, nt*16 + loc) --
#pragma unroll
    for (int mt = 0; mt < 2; ++mt) {
#pragma unroll
        for (int nt = 0; nt < 4; ++nt) {
            const int n = n0 + nt*16 + loc;
#pragma unroll
            for (int j = 0; j < 8; ++j) {
                const int m = m0 + mt*16 + j + 8*half;
                float v = acc[mt][nt][j];
                if constexpr (EPI != EPI_BF16_NOBIAS) v += bias[n];
                if constexpr (EPI == EPI_F32_BIAS_RES) v += res[coff + (size_t)m*ldc + n];
                if constexpr (EPI == EPI_BF16_BIAS_GEL)
                    v = 0.5f * v * (1.0f + erff(v * 0.70710678118654752f));
                if constexpr (EPI == EPI_F32_BIAS || EPI == EPI_F32_BIAS_RES) {
                    ((float*)Cout)[coff + (size_t)m*ldc + n] = v;
                } else {
                    ((unsigned short*)Cout)[coff + (size_t)m*ldc + n] = f2bf(v);
                }
            }
        }
    }
}

// ---------------------------------------------------------------------------
// Patch preprocessing: one wave (32 lanes) per patch.
// ---------------------------------------------------------------------------
__global__ __launch_bounds__(32) void process_patches(
    const float* __restrict__ patches, const float* __restrict__ vocab,
    const float* __restrict__ masked_param, const float* __restrict__ partial_params,
    const int* __restrict__ partial_idx,
    float* __restrict__ processed, int* __restrict__ tokens)
{
    const int i = blockIdx.x;
    const int lane = threadIdx.x;

    float p[16];
#pragma unroll
    for (int d = 0; d < 16; ++d) p[d] = patches[(size_t)i*16 + d];

    int nmask = 0, first = 16;
#pragma unroll
    for (int d = 15; d >= 0; --d) {
        const bool m = fabsf(p[d] - 0.5f) < 0.1f;
        nmask += m;
        if (!m) first = d;
    }
    const bool fully   = (nmask == 16);
    const bool partial = (nmask > 0) && !fully;
    if (first >= 16) first = 0;
    const int val  = (int)rintf(p[first]);
    const int pidx = first*2 + val;
    const bool valid = (pidx >= 0) && (pidx < 32);
    const int  pc    = valid ? pidx : (pidx < 0 ? 0 : 31);

    // argmin_v (||v||^2 - 2 p.v)  (== argmin of squared distance)
    float best = 3.4e38f; int bidx = 0x7fffffff;
    for (int v = lane; v < 8192; v += 32) {
        const float* vr = vocab + (size_t)v*16;
        float dot = 0.f, vv = 0.f;
#pragma unroll
        for (int d = 0; d < 16; ++d) { const float w = vr[d]; dot += w*p[d]; vv += w*w; }
        const float sc = vv - 2.0f*dot;
        if (sc < best || (sc == best && v < bidx)) { best = sc; bidx = v; }
    }
    for (int m = 16; m > 0; m >>= 1) {
        const float ob = __shfl_xor(best, m, 32);
        const int   oi = __shfl_xor(bidx, m, 32);
        if (ob < best || (ob == best && oi < bidx)) { best = ob; bidx = oi; }
    }

    const int tok = fully ? 8192 : (partial ? (valid ? partial_idx[pc] : 0) : bidx);
    if (lane == 0) tokens[i] = tok;
    if (lane < 16) {
        const float pr = fully ? masked_param[lane]
                       : (partial ? (valid ? partial_params[(size_t)pc*16 + lane] : 0.f)
                                  : p[lane]);
        processed[(size_t)i*16 + lane] = pr;
    }
}

// ---------------------------------------------------------------------------
// Token embedding: x[m,h] = processed[m,:] . tok_w[h,:] + tok_b[h]
// ---------------------------------------------------------------------------
__global__ void embed_kernel(const float* __restrict__ proc,
                             const float* __restrict__ tw,
                             const float* __restrict__ tb,
                             float* __restrict__ x)
{
    const int idx = blockIdx.x * blockDim.x + threadIdx.x;
    if (idx >= 4096*768) return;
    const int m = idx / 768, hc = idx % 768;
    float a = tb[hc];
    const float* pr = proc + (size_t)m*16;
    const float* wr = tw   + (size_t)hc*16;
#pragma unroll
    for (int d = 0; d < 16; ++d) a += pr[d]*wr[d];
    x[idx] = a;
}

// ---------------------------------------------------------------------------
// LayerNorm (row of 768) -> bf16, one wave per row
// ---------------------------------------------------------------------------
__global__ __launch_bounds__(32) void ln_bf16(const float* __restrict__ x,
                                              const float* __restrict__ g,
                                              const float* __restrict__ b,
                                              unsigned short* __restrict__ out)
{
    const int row = blockIdx.x, lane = threadIdx.x;
    const float* xr = x + (size_t)row*768;
    float s = 0.f, s2 = 0.f;
    for (int c = lane; c < 768; c += 32) { const float v = xr[c]; s += v; s2 += v*v; }
    for (int m = 16; m > 0; m >>= 1) { s += __shfl_xor(s, m, 32); s2 += __shfl_xor(s2, m, 32); }
    const float mean = s * (1.f/768.f);
    const float var  = s2 * (1.f/768.f) - mean*mean;
    const float rstd = rsqrtf(var + 1e-5f);
    unsigned short* orow = out + (size_t)row*768;
    for (int c = lane; c < 768; c += 32)
        orow[c] = f2bf((xr[c] - mean) * rstd * g[c] + b[c]);
}

// ---------------------------------------------------------------------------
// Attention scores + softmax: one wave per (b, h, s). probs stored bf16.
// q is register-resident (32 packed dwords -> 64 floats); k rows stream as
// contiguous dwords so the compiler can fuse them into b128 loads.
// ---------------------------------------------------------------------------
__global__ __launch_bounds__(32) void attn_scores_softmax(
    const unsigned short* __restrict__ qkvb,  // [4096, 2304] bf16 (q|k|v)
    const float* __restrict__ relavg,         // [65] head-averaged bias, this layer
    unsigned short* __restrict__ probs)       // [B*NH, 256, 256] bf16
{
    const int s = blockIdx.x, h = blockIdx.y, b = blockIdx.z;
    const int lane = threadIdx.x;

    // q row: 64 bf16 = 32 dwords, unpack once into registers
    const unsigned* qp = (const unsigned*)(qkvb + ((size_t)(b*256 + s))*2304 + h*64);
    float q[64];
#pragma unroll
    for (int d = 0; d < 32; ++d) {
        const unsigned u = qp[d];
        q[2*d]   = __uint_as_float(u << 16);
        q[2*d+1] = __uint_as_float(u & 0xffff0000u);
    }

    float sc[8];
#pragma unroll
    for (int it = 0; it < 8; ++it) {
        const int t = it*32 + lane;
        const unsigned* kp =
            (const unsigned*)(qkvb + ((size_t)(b*256 + t))*2304 + 768 + h*64);
        float acc = 0.f;
#pragma unroll
        for (int d = 0; d < 32; ++d) {
            const unsigned u = kp[d];
            acc += q[2*d]   * __uint_as_float(u << 16);
            acc += q[2*d+1] * __uint_as_float(u & 0xffff0000u);
        }
        int rel = s - t;
        rel = rel < -32 ? -32 : (rel > 32 ? 32 : rel);
        sc[it] = acc * 0.125f + relavg[rel + 32];   // 1/sqrt(64)
    }
    float mx = sc[0];
#pragma unroll
    for (int it = 1; it < 8; ++it) mx = fmaxf(mx, sc[it]);
    for (int m = 16; m > 0; m >>= 1) mx = fmaxf(mx, __shfl_xor(mx, m, 32));
    float sum = 0.f;
#pragma unroll
    for (int it = 0; it < 8; ++it) { sc[it] = __expf(sc[it] - mx); sum += sc[it]; }
    for (int m = 16; m > 0; m >>= 1) sum += __shfl_xor(sum, m, 32);
    const float inv = 1.0f / sum;
    unsigned short* prow = probs + (((size_t)(b*12 + h)*256 + s) * 256);
#pragma unroll
    for (int it = 0; it < 8; ++it) prow[it*32 + lane] = f2bf(sc[it] * inv);
}

// ---------------------------------------------------------------------------
// Small utility kernels
// ---------------------------------------------------------------------------
__global__ void f32_to_bf16(const float* __restrict__ in,
                            unsigned short* __restrict__ out, long long n)
{
    long long i = (long long)blockIdx.x * blockDim.x + threadIdx.x;
    const long long stride = (long long)gridDim.x * blockDim.x;
    for (; i < n; i += stride) out[i] = f2bf(in[i]);
}

__global__ void relavg_kernel(const float* __restrict__ relb,  // [L,65,NH]
                              float* __restrict__ out)         // [L,65]
{
    const int idx = blockIdx.x * blockDim.x + threadIdx.x;
    if (idx >= 12*65) return;
    float s = 0.f;
    for (int h = 0; h < 12; ++h) s += relb[(size_t)idx*12 + h];
    out[idx] = s * (1.f/12.f);
}

__global__ void write_tokens(const int* __restrict__ tok, int* __restrict__ dst)
{
    const int i = blockIdx.x * blockDim.x + threadIdx.x;
    if (i < 4096) dst[i] = tok[i];
}

// ---------------------------------------------------------------------------
// Host orchestration
// ---------------------------------------------------------------------------
extern "C" void kernel_launch(void* const* d_in, const int* in_sizes, int n_in,
                              void* d_out, int out_size, void* d_ws, size_t ws_size,
                              hipStream_t stream)
{
    const float* patches        = (const float*)d_in[0];
    const float* vocab          = (const float*)d_in[1];
    const float* masked_param   = (const float*)d_in[2];
    const float* partial_params = (const float*)d_in[3];
    const int*   partial_idx    = (const int*)  d_in[4];
    const float* tok_w  = (const float*)d_in[5];
    const float* tok_b  = (const float*)d_in[6];
    const float* ln1_g  = (const float*)d_in[7];
    const float* ln1_b  = (const float*)d_in[8];
    const float* in_w   = (const float*)d_in[9];
    const float* in_b   = (const float*)d_in[10];
    const float* out_w  = (const float*)d_in[11];
    const float* out_b  = (const float*)d_in[12];
    const float* relb   = (const float*)d_in[13];
    const float* ln2_g  = (const float*)d_in[14];
    const float* ln2_b  = (const float*)d_in[15];
    const float* w1     = (const float*)d_in[16];
    const float* b1     = (const float*)d_in[17];
    const float* w2     = (const float*)d_in[18];
    const float* b2     = (const float*)d_in[19];
    const float* proj_w = (const float*)d_in[20];
    const float* proj_b = (const float*)d_in[21];
    (void)in_sizes; (void)n_in; (void)out_size; (void)ws_size;

    char* ws = (char*)d_ws;
    size_t off = 0;
    auto alloc = [&](size_t bytes) -> char* {
        const size_t o = off;
        off = o + ((bytes + 255) & ~(size_t)255);
        return ws + o;
    };
    // bf16 weight copies (~182 MB total -> fits the 192 MB global L2)
    unsigned short* wq  = (unsigned short*)alloc(12ull*2304*768*2);
    unsigned short* wo  = (unsigned short*)alloc(12ull*768*768*2);
    unsigned short* wm1 = (unsigned short*)alloc(12ull*3072*768*2);
    unsigned short* wm2 = (unsigned short*)alloc(12ull*768*3072*2);
    unsigned short* wp  = (unsigned short*)alloc(8192ull*768*2);
    // activations
    unsigned short* qkvb  = (unsigned short*)alloc(4096ull*2304*2);
    unsigned short* xnb   = (unsigned short*)alloc(4096ull*768*2);
    unsigned short* obuf  = (unsigned short*)alloc(4096ull*768*2);
    unsigned short* probs = (unsigned short*)alloc(192ull*256*256*2);
    unsigned short* hb    = (unsigned short*)alloc(4096ull*3072*2);
    float* xf   = (float*)alloc(4096ull*768*4);
    float* proc = (float*)alloc(4096ull*16*4);
    float* ravg = (float*)alloc(12ull*65*4);
    int*   tok  = (int*)  alloc(4096ull*4);

    auto cvt = [&](const float* src, unsigned short* dst, long long n) {
        const int grid = (int)((n + 255) / 256);
        f32_to_bf16<<<grid, 256, 0, stream>>>(src, dst, n);
    };
    cvt(in_w,   wq,  12LL*2304*768);
    cvt(out_w,  wo,  12LL*768*768);
    cvt(w1,     wm1, 12LL*3072*768);
    cvt(w2,     wm2, 12LL*768*3072);
    cvt(proj_w, wp,  8192LL*768);

    relavg_kernel<<<(12*65 + 255)/256, 256, 0, stream>>>(relb, ravg);
    process_patches<<<4096, 32, 0, stream>>>(patches, vocab, masked_param,
                                             partial_params, partial_idx, proc, tok);
    embed_kernel<<<(4096*768 + 255)/256, 256, 0, stream>>>(proc, tok_w, tok_b, xf);

    for (int l = 0; l < 12; ++l) {
        ln_bf16<<<4096, 32, 0, stream>>>(xf, ln1_g + l*768, ln1_b + l*768, xnb);
        // QKV: [4096,768] x [2304,768]^T -> bf16 [4096,2304]
        gemm_wmma<EPI_BF16_BIAS, true><<<dim3(2304/64, 4096/128, 1), 128, 0, stream>>>(
            xnb, 768, 0, 0,
            wq + (size_t)l*2304*768, 768, 0, 0,
            in_b + (size_t)l*2304, nullptr,
            qkvb, 2304, 0, 0, 768, 1);
        // scores + rel-bias + softmax -> probs bf16
        attn_scores_softmax<<<dim3(256, 12, 16), 32, 0, stream>>>(qkvb, ravg + l*65, probs);
        // A.V: batched over z = b*12 + h, per batch [256,256] x [256,64]
        gemm_wmma<EPI_BF16_NOBIAS, false><<<dim3(1, 256/128, 192), 128, 0, stream>>>(
            probs, 256, 12LL*256*256, 256LL*256,
            qkvb + 1536, 2304, 256LL*2304, 64,
            nullptr, nullptr,
            obuf, 768, 256LL*768, 64, 256, 12);
        // out-proj + residual (f32)
        gemm_wmma<EPI_F32_BIAS_RES, true><<<dim3(768/64, 4096/128, 1), 128, 0, stream>>>(
            obuf, 768, 0, 0,
            wo + (size_t)l*768*768, 768, 0, 0,
            out_b + (size_t)l*768, xf,
            xf, 768, 0, 0, 768, 1);
        ln_bf16<<<4096, 32, 0, stream>>>(xf, ln2_g + l*768, ln2_b + l*768, xnb);
        // MLP up + exact GELU -> bf16 [4096,3072]
        gemm_wmma<EPI_BF16_BIAS_GEL, true><<<dim3(3072/64, 4096/128, 1), 128, 0, stream>>>(
            xnb, 768, 0, 0,
            wm1 + (size_t)l*3072*768, 768, 0, 0,
            b1 + (size_t)l*3072, nullptr,
            hb, 3072, 0, 0, 768, 1);
        // MLP down + residual (f32)
        gemm_wmma<EPI_F32_BIAS_RES, true><<<dim3(768/64, 4096/128, 1), 128, 0, stream>>>(
            hb, 3072, 0, 0,
            wm2 + (size_t)l*768*3072, 3072, 0, 0,
            b2 + (size_t)l*768, xf,
            xf, 768, 0, 0, 3072, 1);
    }

    // final projection -> logits f32 [4096, 8192] straight into d_out
    f32_to_bf16<<<(4096*768 + 255)/256, 256, 0, stream>>>(xf, xnb, 4096LL*768);
    gemm_wmma<EPI_F32_BIAS, true><<<dim3(8192/64, 4096/128, 1), 128, 0, stream>>>(
        xnb, 768, 0, 0,
        wp, 768, 0, 0,
        proj_b, nullptr,
        (float*)d_out, 8192, 0, 0, 768, 1);

    // tokens appended after logits
    write_tokens<<<(4096 + 255)/256, 256, 0, stream>>>(tok, (int*)d_out + 4096LL*8192);
}
